// AllegroHaiku_8933531975691
// MI455X (gfx1250) — matrix operand order repaired and checked by reading, compiled, not run on hardware
//
#include <hip/hip_runtime.h>

#define E_EDGES 131072
#define N_NODES 8192
#define HDIM    256
#define MULT    16
#define FDIM    16

typedef __attribute__((ext_vector_type(16))) __bf16 v16bf;
typedef __attribute__((ext_vector_type(8)))  __bf16 v8bf;
typedef __attribute__((ext_vector_type(8)))  float  v8f;
typedef __attribute__((ext_vector_type(2)))  float  v2f;

__device__ __forceinline__ float silu_f(float x) { return x / (1.f + __expf(-x)); }

// ---------------------------------------------------------------------------
// Pack f32 weight matrix (Kraw x Ncols) into WMMA bf16 B-fragment layout.
// Fragment layout per 32-K x 16-N tile (kt, nt):
//   bf16 index = ((kt*NT + nt)*32 + lane)*16 + j
//   n = nt*16 + (lane & 15)
//   k = kt*32 + (lane>>4)*8 + (j&7) + (j>>3)*16   (lanes 0-15: K 0..7,16..23, lanes 16-31: K 8..15,24..31)
// K zero-padded to KT*32.
// ---------------------------------------------------------------------------
__global__ void pack_weights(const float* __restrict__ W, __bf16* __restrict__ out,
                             int Kraw, int KT, int Ncols, int NT) {
    int gid = blockIdx.x * 256 + threadIdx.x;
    int total = KT * NT * 512;
    if (gid >= total) return;
    int j    = gid & 15;
    int lane = (gid >> 4) & 31;
    int tile = gid >> 9;
    int nt   = tile % NT;
    int kt   = tile / NT;
    int n = nt * 16 + (lane & 15);
    int k = kt * 32 + (lane >> 4) * 8 + (j & 7) + (j >> 3) * 16;
    float v = (k < Kraw) ? W[(size_t)k * Ncols + n] : 0.f;
    out[gid] = (__bf16)v;
}

// ---------------------------------------------------------------------------
// Edge featurization: envelope u, Bessel basis + gathered node attrs -> e0
// (bf16, 64-wide, zero padded), spherical harmonics Y (f32, 16), and zero the
// tail (cols 256..287) of the 288-wide bf16 activation buffer hb.
// ---------------------------------------------------------------------------
__global__ void edge_features(const float* __restrict__ vectors,
                              const float* __restrict__ node_attrs,
                              const int* __restrict__ senders,
                              const int* __restrict__ receivers,
                              float* __restrict__ uArr,
                              float* __restrict__ Y,
                              __bf16* __restrict__ e0,
                              __bf16* __restrict__ hb,
                              int E) {
    int e = blockIdx.x * 256 + threadIdx.x;
    if (e >= E) return;
    float vx = vectors[3 * e + 0], vy = vectors[3 * e + 1], vz = vectors[3 * e + 2];
    float d = sqrtf(vx * vx + vy * vy + vz * vz);

    // envelope: p=6, cutoff=1 -> 1 - 28 d^6 + 48 d^7 - 21 d^8
    float d2 = d * d, d3 = d2 * d, d6 = d3 * d3, d7 = d6 * d, d8 = d7 * d;
    float uu = 1.f - 28.f * d6 + 48.f * d7 - 21.f * d8;
    uu = (d < 1.f) ? uu : 0.f;
    uArr[e] = uu;

    // Bessel basis: sqrt(2)*sin(k*pi*d)/d, k=1..8
    const float PI_F = 3.14159265358979f;
    const float S2   = 1.41421356237f;
    __bf16* erow = e0 + (size_t)e * 64;
    float rd = 1.f / d;
#pragma unroll
    for (int k = 1; k <= 8; ++k)
        erow[k - 1] = (__bf16)(S2 * __sinf((float)k * PI_F * d) * rd);

    int s = senders[e], r = receivers[e];
    const float* na = node_attrs + (size_t)s * FDIM;
    const float* nb = node_attrs + (size_t)r * FDIM;
#pragma unroll
    for (int i = 0; i < FDIM; ++i) erow[8 + i]  = (__bf16)na[i];
#pragma unroll
    for (int i = 0; i < FDIM; ++i) erow[24 + i] = (__bf16)nb[i];
#pragma unroll
    for (int i = 40; i < 64; ++i) erow[i] = (__bf16)0.f;

    // spherical harmonics on unit vector
    float x = vx * rd, y = vy * rd, z = vz * rd;
    const float s3 = 1.73205080757f, s5 = 2.23606797750f, s15 = 3.87298334621f;
    const float s7 = 2.64575131106f, c33 = 2.09165006634f, c32 = 10.2469507660f;
    const float c31 = 1.62018517460f;
    float* yr = Y + (size_t)e * 16;
    yr[0]  = 1.f;
    yr[1]  = s3 * x;
    yr[2]  = s3 * y;
    yr[3]  = s3 * z;
    yr[4]  = s15 * x * y;
    yr[5]  = s15 * y * z;
    yr[6]  = 0.5f * s5 * (3.f * z * z - 1.f);
    yr[7]  = s15 * x * z;
    yr[8]  = 0.5f * s15 * (x * x - y * y);
    yr[9]  = c33 * y * (3.f * x * x - y * y);
    yr[10] = c32 * x * y * z;
    yr[11] = c31 * y * (5.f * z * z - 1.f);
    yr[12] = 0.5f * s7 * (5.f * z * z * z - 3.f * z);
    yr[13] = c31 * x * (5.f * z * z - 1.f);
    yr[14] = 0.5f * c32 * z * (x * x - y * y);
    yr[15] = c33 * x * (x * x - 3.f * y * y);

    // zero hb tail cols 256..287 (VV0 slot written per layer; pad stays 0)
    __bf16* ht = hb + (size_t)e * 288 + 256;
#pragma unroll
    for (int i = 0; i < 32; ++i) ht[i] = (__bf16)0.f;
}

// ---------------------------------------------------------------------------
// WMMA GEMM with register blocking. C[rows x NTOT*16] = A[rows x KT*32] * B.
// Wave tile: MT row-tiles (MT*16 rows) x NTW col-tiles (NTW*16 cols).
// Block: 8 waves arranged as (8/COLGRPS) row-groups x COLGRPS col-groups,
// COLGRPS = NTOT/NTW. Each B fragment is reused by MT WMMAs and each A
// fragment by NTW WMMAs -> ~4x less cache traffic per v_wmma at MT=NTW=4.
// A: row-major bf16 (lda). A-fragment per documented 16-bit 16x32 layout:
// lane<16 holds K {base..base+7, base+16..base+23}, base = khalf*8.
// B: packed fragments (pack_weights layout).
// Epilogue (compile-time): +bias, SiLU, *u[row], residual (x+u*y)/sqrt(2),
// write f32 and/or bf16.
// ---------------------------------------------------------------------------
template <int KT, int NTOT, int MT, int NTW, bool BIAS, bool SILU, bool SCALEU,
          bool RESID, bool WF32, bool WBF16>
__global__ __launch_bounds__(256) void gemm_wmma(
    const __bf16* __restrict__ A, int lda,
    const __bf16* __restrict__ Bp,
    const float* __restrict__ bias,
    float* __restrict__ Cf, int ldcf,
    __bf16* __restrict__ Cb, int ldcb,
    const float* __restrict__ uvec,
    const float* __restrict__ resid, int ldres,
    int nrows) {
    constexpr int COLGRPS = NTOT / NTW;
    constexpr int ROWGRPS = 8 / COLGRPS;
    constexpr int ROWS_PER_BLOCK = ROWGRPS * MT * 16;

    int w = threadIdx.x >> 5;
    int lane = threadIdx.x & 31;
    int colgrp = w % COLGRPS;
    int rowgrp = w / COLGRPS;
    int rowBase = blockIdx.x * ROWS_PER_BLOCK + rowgrp * MT * 16;
    if (rowBase >= nrows) return;
    int m = lane & 15;
    int khalf = lane >> 4;

    v8f acc[MT][NTW] = {};
    const __bf16* arow0 = A + (size_t)(rowBase + m) * lda + khalf * 8;
    const v16bf* bbase = (const v16bf*)Bp;

    for (int kt = 0; kt < KT; ++kt) {
        // prefetch next K-slab of the streaming A rows (global_prefetch_b8)
        if (kt + 1 < KT) __builtin_prefetch(arow0 + (kt + 1) * 32, 0, 1);

        v16bf a[MT];
#pragma unroll
        for (int mt = 0; mt < MT; ++mt) {
            const __bf16* ap = arow0 + (size_t)(mt * 16) * lda + kt * 32;
            v8bf lo = *(const v8bf*)(ap);
            v8bf hi = *(const v8bf*)(ap + 16);
            a[mt] = __builtin_shufflevector(lo, hi, 0, 1, 2, 3, 4, 5, 6, 7, 8,
                                            9, 10, 11, 12, 13, 14, 15);
        }
#pragma unroll
        for (int nt = 0; nt < NTW; ++nt) {
            v16bf b = bbase[(size_t)(kt * NTOT + colgrp * NTW + nt) * 32 + lane];
#pragma unroll
            for (int mt = 0; mt < MT; ++mt) {
                acc[mt][nt] = __builtin_amdgcn_wmma_f32_16x16x32_bf16(
                    false, a[mt], false, b, (short)0, acc[mt][nt], false, false);
            }
        }
    }

#pragma unroll
    for (int mt = 0; mt < MT; ++mt) {
#pragma unroll
        for (int nt = 0; nt < NTW; ++nt) {
            int col = (colgrp * NTW + nt) * 16 + m;
            float bcol = BIAS ? bias[col] : 0.f;
#pragma unroll
            for (int v = 0; v < 8; ++v) {
                int row = rowBase + mt * 16 + khalf * 8 + v;
                float x = acc[mt][nt][v];
                if (BIAS)   x += bcol;
                if (SILU)   x = silu_f(x);
                if (SCALEU) x *= uvec[row];
                if (RESID)  x = (resid[(size_t)row * ldres + col] +
                                 uvec[row] * x) * 0.70710678118f;
                if (WF32)  Cf[(size_t)row * ldcf + col] = x;
                if (WBF16) Cb[(size_t)row * ldcb + col] = (__bf16)x;
            }
        }
    }
}

// V[e][m][i] = vw[e][m] * Y[e][i]
__global__ void init_V(const float* __restrict__ vw, const float* __restrict__ Y,
                       float* __restrict__ V, int total) {
    int g = blockIdx.x * 256 + threadIdx.x;
    if (g >= total) return;
    int e = g >> 4;
    float vm = vw[g];
    const float* yr = Y + (size_t)e * 16;
    float* out = V + (size_t)g * 16;
#pragma unroll
    for (int i = 0; i < 16; ++i) out[i] = vm * yr[i];
}

// wY[senders[e]][m][i] += w[e][m]*Y[e][i]
__global__ void scatter_wY(const float* __restrict__ vw, const float* __restrict__ Y,
                           const int* __restrict__ senders, float* __restrict__ wY,
                           int total) {
    int g = blockIdx.x * 256 + threadIdx.x;
    if (g >= total) return;
    int e = g >> 4, mm = g & 15;
    float wm = vw[g];
    int s = senders[e];
    const float* yr = Y + (size_t)e * 16;
    float* dst = wY + (size_t)s * 256 + mm * 16;
#pragma unroll
    for (int i = 0; i < 16; ++i) atomicAdd(dst + i, wm * yr[i]);
}

// ---------------------------------------------------------------------------
// Per-edge tensor-product layer core. One wave per edge:
//   VV = wY[sender]*inv * V   (16x16)
//   V  = VV @ W_lsh           (16x16x16, via v_wmma_f32_16x16x4_f32)
//   hb[e][256+m] = VV[m][0]   (bf16)
// ---------------------------------------------------------------------------
__global__ __launch_bounds__(256) void layer_edge(
    const int* __restrict__ senders, const float* __restrict__ wY,
    float* __restrict__ V, const float* __restrict__ Wlsh,
    __bf16* __restrict__ hb, int E) {
    int wid = (blockIdx.x * 256 + threadIdx.x) >> 5;
    int lane = threadIdx.x & 31;
    if (wid >= E) return;
    int e = wid;
    int m = lane & 15, kh = lane >> 4;
    int s = senders[e];
    const float* wYr = wY + (size_t)s * 256 + m * 16;
    const float* Vr  = V + (size_t)e * 256 + m * 16;
    const float inv = 0.25f;  // 1/sqrt(AVG_NEIGH=16)

    float vv[8];
#pragma unroll
    for (int kt = 0; kt < 4; ++kt) {
        int k0 = kt * 4 + kh * 2;
        vv[kt * 2]     = wYr[k0] * inv * Vr[k0];
        vv[kt * 2 + 1] = wYr[k0 + 1] * inv * Vr[k0 + 1];
    }

#if __has_builtin(__builtin_amdgcn_wmma_f32_16x16x4_f32)
    v8f c = {};
#pragma unroll
    for (int kt = 0; kt < 4; ++kt) {
        int k0 = kt * 4 + kh * 2;
        v2f a = {vv[kt * 2], vv[kt * 2 + 1]};
        v2f b = {Wlsh[k0 * 16 + m], Wlsh[(k0 + 1) * 16 + m]};
        c = __builtin_amdgcn_wmma_f32_16x16x4_f32(false, a, false, b, (short)0,
                                                  c, false, false);
    }
    float* Vo = V + (size_t)e * 256;
#pragma unroll
    for (int v = 0; v < 8; ++v) Vo[(kh * 8 + v) * 16 + m] = c[v];
#else
    // VALU fallback: each lane owns outputs (row = kh*8+v, col = m)
    float out[8];
#pragma unroll
    for (int v = 0; v < 8; ++v) {
        int row = kh * 8 + v;
        float sum = 0.f;
#pragma unroll
        for (int i = 0; i < 16; ++i) {
            float vvri = wY[(size_t)s * 256 + row * 16 + i] * inv *
                         V[(size_t)e * 256 + row * 16 + i];
            sum += vvri * Wlsh[i * 16 + m];
        }
        out[v] = sum;
    }
#pragma unroll
    for (int v = 0; v < 8; ++v) V[(size_t)e * 256 + (kh * 8 + v) * 16 + m] = out[v];
#endif

    if (kh == 0) hb[(size_t)e * 288 + 256 + m] = (__bf16)vv[0];  // VV[m][0]
}

// edge_out = (x . W_out) * u ; node_out[recv] += edge_out * inv
__global__ __launch_bounds__(256) void edge_out_scatter(
    const float* __restrict__ xf, const float* __restrict__ Wout,
    const float* __restrict__ uArr, const int* __restrict__ receivers,
    float* __restrict__ out, int E) {
    int wid = (blockIdx.x * 256 + threadIdx.x) >> 5;
    int lane = threadIdx.x & 31;
    if (wid >= E) return;
    const float* xr = xf + (size_t)wid * 256;
    float s = 0.f;
#pragma unroll
    for (int j = 0; j < 8; ++j) {
        int c = lane + 32 * j;
        s += xr[c] * Wout[c];
    }
    for (int o = 16; o > 0; o >>= 1) s += __shfl_xor(s, o, 32);
    if (lane == 0) atomicAdd(out + receivers[wid], s * uArr[wid] * 0.25f);
}

// ---------------------------------------------------------------------------
extern "C" void kernel_launch(void* const* d_in, const int* in_sizes, int n_in,
                              void* d_out, int out_size, void* d_ws, size_t ws_size,
                              hipStream_t stream) {
    (void)in_sizes; (void)n_in; (void)ws_size;
    const int E = E_EDGES, N = N_NODES;

    const float* node_attrs = (const float*)d_in[0];
    const float* vectors    = (const float*)d_in[1];
    const int*   senders    = (const int*)d_in[2];
    const int*   receivers  = (const int*)d_in[3];
    const float* W_e0  = (const float*)d_in[4];
    const float* b_e0  = (const float*)d_in[5];
    const float* W_e1  = (const float*)d_in[6];
    const float* b_e1  = (const float*)d_in[7];
    const float* W_v0  = (const float*)d_in[8];
    const float* W_lw  = (const float*)d_in[9];
    const float* W_lsh = (const float*)d_in[10];
    const float* W_ly1 = (const float*)d_in[11];
    const float* b_ly1 = (const float*)d_in[12];
    const float* W_ly2 = (const float*)d_in[13];
    const float* b_ly2 = (const float*)d_in[14];
    const float* W_out = (const float*)d_in[15];

    char* ws = (char*)d_ws;
    size_t off = 0;
    auto alloc = [&](size_t bytes) -> char* {
        char* p = ws + off;
        off += (bytes + 255) & ~(size_t)255;
        return p;
    };

    float*  uArr = (float*)alloc((size_t)E * 4);
    float*  Y    = (float*)alloc((size_t)E * 16 * 4);
    __bf16* e0   = (__bf16*)alloc((size_t)E * 64 * 2);
    __bf16* hb   = (__bf16*)alloc((size_t)E * 288 * 2);
    float*  xf   = (float*)alloc((size_t)E * 256 * 4);
    float*  V    = (float*)alloc((size_t)E * 256 * 4);
    float*  vw   = (float*)alloc((size_t)E * 16 * 4);
    float*  wYb  = (float*)alloc((size_t)N * 256 * 4);
    __bf16* pWe0    = (__bf16*)alloc(2 * 16 * 512 * 2);
    __bf16* pWe1    = (__bf16*)alloc(8 * 16 * 512 * 2);
    __bf16* pWv0    = (__bf16*)alloc(8 * 1 * 512 * 2);
    __bf16* pWlw[2]  = {(__bf16*)alloc(8 * 1 * 512 * 2),  (__bf16*)alloc(8 * 1 * 512 * 2)};
    __bf16* pWly1[2] = {(__bf16*)alloc(9 * 16 * 512 * 2), (__bf16*)alloc(9 * 16 * 512 * 2)};
    __bf16* pWly2[2] = {(__bf16*)alloc(8 * 16 * 512 * 2), (__bf16*)alloc(8 * 16 * 512 * 2)};

    auto pack = [&](const float* W, __bf16* o, int Kraw, int KT, int Ncols, int NT) {
        int total = KT * NT * 512;
        pack_weights<<<total / 256, 256, 0, stream>>>(W, o, Kraw, KT, Ncols, NT);
    };
    pack(W_e0, pWe0, 40, 2, 256, 16);
    pack(W_e1, pWe1, 256, 8, 256, 16);
    pack(W_v0, pWv0, 256, 8, 16, 1);
    for (int l = 0; l < 2; ++l) {
        pack(W_lw + (size_t)l * 256 * 16, pWlw[l], 256, 8, 16, 1);
        pack(W_ly1 + (size_t)l * 272 * 256, pWly1[l], 272, 9, 256, 16);
        pack(W_ly2 + (size_t)l * 256 * 256, pWly2[l], 256, 8, 256, 16);
    }

    edge_features<<<E / 256, 256, 0, stream>>>(vectors, node_attrs, senders,
                                               receivers, uArr, Y, e0, hb, E);

    const int GB_WIDE = E / 128;  // 128 rows/block (2 rowgrps x 4 colgrps, 64x64 wave tile)
    const int GB_THIN = E / 512;  // 512 rows/block (8 rowgrps, N=16)

    // x = silu(e0 @ W_e0 + b_e0)  -> hb (bf16, stride 288)
    gemm_wmma<2, 16, 4, 4, true, true, false, false, false, true>
        <<<GB_WIDE, 256, 0, stream>>>(e0, 64, pWe0, b_e0, nullptr, 0, hb, 288,
                                      nullptr, nullptr, 0, E);
    // x = u * silu(x @ W_e1 + b_e1) -> xf (f32) + hb (bf16)
    gemm_wmma<8, 16, 4, 4, true, true, true, false, true, true>
        <<<GB_WIDE, 256, 0, stream>>>(hb, 288, pWe1, b_e1, xf, 256, hb, 288,
                                      uArr, nullptr, 0, E);
    // v = x @ W_v0 -> vw (f32, E x 16)
    gemm_wmma<8, 1, 4, 1, false, false, false, false, true, false>
        <<<GB_THIN, 256, 0, stream>>>(hb, 288, pWv0, nullptr, vw, 16, nullptr,
                                      0, nullptr, nullptr, 0, E);
    init_V<<<E * 16 / 256, 256, 0, stream>>>(vw, Y, V, E * 16);

    for (int l = 0; l < 2; ++l) {
        // w = x @ W_lw[l]
        gemm_wmma<8, 1, 4, 1, false, false, false, false, true, false>
            <<<GB_THIN, 256, 0, stream>>>(hb, 288, pWlw[l], nullptr, vw, 16,
                                          nullptr, 0, nullptr, nullptr, 0, E);
        hipMemsetAsync(wYb, 0, (size_t)N * 256 * 4, stream);
        scatter_wY<<<E * 16 / 256, 256, 0, stream>>>(vw, Y, senders, wYb, E * 16);
        layer_edge<<<E / 8, 256, 0, stream>>>(senders, wYb, V,
                                              W_lsh + (size_t)l * 256, hb, E);
        // y = silu(h @ W_ly1 + b)  (reads K=288 incl. VV0 cols, writes cols 0..255)
        gemm_wmma<9, 16, 4, 4, true, true, false, false, false, true>
            <<<GB_WIDE, 256, 0, stream>>>(hb, 288, pWly1[l], b_ly1 + l * 256,
                                          nullptr, 0, hb, 288, nullptr,
                                          nullptr, 0, E);
        // y = silu(y @ W_ly2 + b); x = (x + u*y)/sqrt(2) -> xf + hb
        gemm_wmma<8, 16, 4, 4, true, true, false, true, true, true>
            <<<GB_WIDE, 256, 0, stream>>>(hb, 288, pWly2[l], b_ly2 + l * 256,
                                          xf, 256, hb, 288, uArr, xf, 256, E);
    }

    hipMemsetAsync(d_out, 0, (size_t)out_size * sizeof(float), stream);
    edge_out_scatter<<<E / 8, 256, 0, stream>>>(xf, W_out, uArr, receivers,
                                                (float*)d_out, E);
}
